// CRF_Layer_50139448213609
// MI455X (gfx1250) — matrix operand order repaired and checked by reading, not compiled, not run on hardware
//
#include <hip/hip_runtime.h>
#include <hip/hip_bf16.h>

typedef __attribute__((ext_vector_type(16))) _Float16 v16h;
typedef __attribute__((ext_vector_type(8)))  float    v8f;

#define NEG_INF (-1.0e30f)

static constexpr int   MSEQ   = 200000;   // sequence length
static constexpr int   DFEAT  = 128;      // feature dim
static constexpr int   KLAB   = 26;       // labels
static constexpr int   NCHUNK = 1024;     // DP chunks
static constexpr int   CHUNKL = (MSEQ - 1 + NCHUNK - 1) / NCHUNK;  // 196

// ---------------------------------------------------------------------------
// K1: emission scores = x @ w^T via v_wmma_f32_16x16x32_f16.
// One wave per 16-row tile; N covered by two 16-label tiles (26 -> pad 32).
// x read exactly once (HBM-bound floor ~4.4us at 23.3 TB/s).
// scores stored [M][32] f32 (cols 26..31 garbage, never read by DP).
// ---------------------------------------------------------------------------
__global__ __launch_bounds__(128) void k_scores(const float* __restrict__ x,
                                                const float* __restrict__ w,
                                                float* __restrict__ scores) {
    const int wave = threadIdx.x >> 5;
    const int lane = threadIdx.x & 31;
    const int tile = blockIdx.x * 4 + wave;   // 12500 tiles of 16 rows
    const int lrow = lane & 15;
    const int grp  = lane >> 4;
    const int row  = tile * 16 + lrow;

    const float* xrow = x + (size_t)row * DFEAT;
    const int lab0 = lrow;
    const int lab1 = 16 + lrow;
    const bool lab1ok = (lab1 < KLAB);
    const float* wrow0 = w + (size_t)lab0 * DFEAT;
    const float* wrow1 = w + (size_t)(lab1ok ? lab1 : 0) * DFEAT;

    v8f c0 = {};
    v8f c1 = {};

    #pragma unroll
    for (int ch = 0; ch < 4; ++ch) {
        const int kb = ch * 32;
        v16h a, b0, b1;
        // A 16x32 f16 layout: lane group g holds K = kb + g*8 + 0..7 (v0..3)
        // and K = kb + 16 + g*8 + 0..7 (v4..7), row M = lrow.
        {
            const float4 f0 = *(const float4*)(xrow + kb + grp * 8);
            const float4 f1 = *(const float4*)(xrow + kb + grp * 8 + 4);
            const float4 f2 = *(const float4*)(xrow + kb + 16 + grp * 8);
            const float4 f3 = *(const float4*)(xrow + kb + 16 + grp * 8 + 4);
            a[0]=(_Float16)f0.x;  a[1]=(_Float16)f0.y;  a[2]=(_Float16)f0.z;  a[3]=(_Float16)f0.w;
            a[4]=(_Float16)f1.x;  a[5]=(_Float16)f1.y;  a[6]=(_Float16)f1.z;  a[7]=(_Float16)f1.w;
            a[8]=(_Float16)f2.x;  a[9]=(_Float16)f2.y;  a[10]=(_Float16)f2.z; a[11]=(_Float16)f2.w;
            a[12]=(_Float16)f3.x; a[13]=(_Float16)f3.y; a[14]=(_Float16)f3.z; a[15]=(_Float16)f3.w;
        }
        // B 32x16 f16 layout: lane n (=lrow) in group g holds
        // B[K = g*8 + 0..7 and 16 + g*8 + 0..7][n]; B[k][n] = w[label n][kb + k].
        {
            const float4 g0 = *(const float4*)(wrow0 + kb + grp * 8);
            const float4 g1 = *(const float4*)(wrow0 + kb + grp * 8 + 4);
            const float4 g2 = *(const float4*)(wrow0 + kb + 16 + grp * 8);
            const float4 g3 = *(const float4*)(wrow0 + kb + 16 + grp * 8 + 4);
            b0[0]=(_Float16)g0.x;  b0[1]=(_Float16)g0.y;  b0[2]=(_Float16)g0.z;  b0[3]=(_Float16)g0.w;
            b0[4]=(_Float16)g1.x;  b0[5]=(_Float16)g1.y;  b0[6]=(_Float16)g1.z;  b0[7]=(_Float16)g1.w;
            b0[8]=(_Float16)g2.x;  b0[9]=(_Float16)g2.y;  b0[10]=(_Float16)g2.z; b0[11]=(_Float16)g2.w;
            b0[12]=(_Float16)g3.x; b0[13]=(_Float16)g3.y; b0[14]=(_Float16)g3.z; b0[15]=(_Float16)g3.w;
        }
        {
            const float4 g0 = *(const float4*)(wrow1 + kb + grp * 8);
            const float4 g1 = *(const float4*)(wrow1 + kb + grp * 8 + 4);
            const float4 g2 = *(const float4*)(wrow1 + kb + 16 + grp * 8);
            const float4 g3 = *(const float4*)(wrow1 + kb + 16 + grp * 8 + 4);
            const _Float16 msk = lab1ok ? (_Float16)1.0f : (_Float16)0.0f;
            b1[0]=msk*(_Float16)g0.x;  b1[1]=msk*(_Float16)g0.y;  b1[2]=msk*(_Float16)g0.z;  b1[3]=msk*(_Float16)g0.w;
            b1[4]=msk*(_Float16)g1.x;  b1[5]=msk*(_Float16)g1.y;  b1[6]=msk*(_Float16)g1.z;  b1[7]=msk*(_Float16)g1.w;
            b1[8]=msk*(_Float16)g2.x;  b1[9]=msk*(_Float16)g2.y;  b1[10]=msk*(_Float16)g2.z; b1[11]=msk*(_Float16)g2.w;
            b1[12]=msk*(_Float16)g3.x; b1[13]=msk*(_Float16)g3.y; b1[14]=msk*(_Float16)g3.z; b1[15]=msk*(_Float16)g3.w;
        }
        c0 = __builtin_amdgcn_wmma_f32_16x16x32_f16(false, a, false, b0, (short)0, c0, false, false);
        c1 = __builtin_amdgcn_wmma_f32_16x16x32_f16(false, a, false, b1, (short)0, c1, false, false);
    }

    // C layout: VGPR r, lanes 0-15 -> (M=r, N=lane); lanes 16-31 -> (M=8+r, N=lane-16)
    const int rbase = tile * 16 + grp * 8;
    #pragma unroll
    for (int r = 0; r < 8; ++r) {
        scores[(size_t)(rbase + r) * 32 + lrow]      = c0[r];
        scores[(size_t)(rbase + r) * 32 + 16 + lrow] = c1[r];
    }
}

// ---------------------------------------------------------------------------
// K2: per-chunk max-plus transition matrix R_c (26x26).
// Step op A_i[j][k] = T[j][k] + s_i[k]; fold: R'[j][k] = max_l(R[j][l]+T[l][k]) + s_i[k].
// One wave per chunk; lane k owns column k; R double-buffered in LDS.
// ---------------------------------------------------------------------------
__global__ __launch_bounds__(32) void k_chunkmat(const float* __restrict__ scores,
                                                 const float* __restrict__ T,
                                                 float* __restrict__ Rg) {
    const int c = blockIdx.x;
    const int k = threadIdx.x;
    const bool act = (k < KLAB);
    const long beg = (long)c * CHUNKL;
    long end = beg + CHUNKL; if (end > MSEQ - 1) end = MSEQ - 1;

    __shared__ float R[2][KLAB][KLAB];
    float Tcol[KLAB];
    #pragma unroll
    for (int j = 0; j < KLAB; ++j) Tcol[j] = act ? T[j * KLAB + k] : NEG_INF;

    int cur = 0;
    if (beg + 1 <= end) {
        const float s = act ? scores[(size_t)(beg + 1) * 32 + k] : 0.0f;
        if (act) {
            #pragma unroll
            for (int j = 0; j < KLAB; ++j) R[0][j][k] = Tcol[j] + s;
        }
    } else {  // empty chunk -> max-plus identity
        if (act) {
            #pragma unroll
            for (int j = 0; j < KLAB; ++j) R[0][j][k] = (j == k) ? 0.0f : NEG_INF;
        }
    }
    __syncthreads();

    for (long i = beg + 2; i <= end; ++i) {
        const float s = act ? scores[(size_t)i * 32 + k] : 0.0f;
        #pragma unroll 1
        for (int j = 0; j < KLAB; ++j) {
            float m = NEG_INF;
            #pragma unroll
            for (int l = 0; l < KLAB; ++l) m = fmaxf(m, R[cur][j][l] + Tcol[l]);
            if (act) R[1 - cur][j][k] = m + s;   // write other buffer: no race
        }
        cur ^= 1;
        __syncthreads();
    }

    if (act) {
        #pragma unroll
        for (int j = 0; j < KLAB; ++j)
            Rg[(size_t)c * (KLAB * KLAB) + j * KLAB + k] = R[cur][j][k];
    }
}

// ---------------------------------------------------------------------------
// K3: single-wave scan over chunk matrices -> exact alpha at each chunk start,
// plus final argmax label (first-index tie-break like jnp.argmax).
// ---------------------------------------------------------------------------
__global__ __launch_bounds__(32) void k_scan(const float* __restrict__ scores,
                                             const float* __restrict__ Rg,
                                             float* __restrict__ alphaIn,
                                             int* __restrict__ lastOut,
                                             int* __restrict__ path) {
    const int k = threadIdx.x;
    const bool act = (k < KLAB);
    float v = act ? scores[k] : NEG_INF;   // alpha_0 = scores[0]
    for (int c = 0; c < NCHUNK; ++c) {
        if (act) alphaIn[(size_t)c * 32 + k] = v;
        const float* Rc = Rg + (size_t)c * (KLAB * KLAB);
        float m = NEG_INF;
        #pragma unroll
        for (int j = 0; j < KLAB; ++j) {
            const float vj = __shfl(v, j, 32);
            const float r  = act ? Rc[j * KLAB + k] : NEG_INF;
            m = fmaxf(m, vj + r);
        }
        v = act ? m : NEG_INF;
    }
    // argmax over lanes 0..25 (strict > keeps first index)
    float best = __shfl(v, 0, 32);
    int bi = 0;
    #pragma unroll
    for (int j = 1; j < KLAB; ++j) {
        const float vj = __shfl(v, j, 32);
        if (vj > best) { best = vj; bi = j; }
    }
    if (threadIdx.x == 0) { *lastOut = bi; path[MSEQ - 1] = bi; }
}

// ---------------------------------------------------------------------------
// K4: per-chunk exact DP from known alpha_in: store backpointers (global +
// LDS), then backtrack all 26 end labels inside LDS -> end->start map g_c.
// ---------------------------------------------------------------------------
__global__ __launch_bounds__(32) void k_backptr(const float* __restrict__ scores,
                                                const float* __restrict__ T,
                                                const float* __restrict__ alphaIn,
                                                unsigned char* __restrict__ bp,
                                                unsigned char* __restrict__ g) {
    const int c = blockIdx.x;
    const int k = threadIdx.x;
    const bool act = (k < KLAB);
    const long beg = (long)c * CHUNKL;
    long end = beg + CHUNKL; if (end > MSEQ - 1) end = MSEQ - 1;

    __shared__ unsigned char bpS[CHUNKL][32];
    float Tcol[KLAB];
    #pragma unroll
    for (int j = 0; j < KLAB; ++j) Tcol[j] = act ? T[j * KLAB + k] : NEG_INF;

    float alpha = act ? alphaIn[(size_t)c * 32 + k] : NEG_INF;
    for (long i = beg + 1; i <= end; ++i) {
        float m = NEG_INF; int bi = 0;
        #pragma unroll
        for (int j = 0; j < KLAB; ++j) {
            const float cand = __shfl(alpha, j, 32) + Tcol[j];
            if (cand > m) { m = cand; bi = j; }
        }
        alpha = act ? (m + scores[(size_t)i * 32 + k]) : NEG_INF;
        if (act) {
            bpS[i - beg - 1][k] = (unsigned char)bi;
            bp[(size_t)i * 32 + k] = (unsigned char)bi;
        }
    }
    __syncthreads();
    if (act) {
        int curl = k;
        for (long t = end - beg - 1; t >= 0; --t) curl = bpS[t][curl];
        g[(size_t)c * 32 + k] = (unsigned char)curl;   // identity for empty chunks
    }
}

// ---------------------------------------------------------------------------
// K5: compose chunk maps -> true label at every chunk boundary.
// All g maps staged in LDS (32KB of the 320KB WGP LDS) so the 1024-long
// dependent chain runs at LDS latency.
// ---------------------------------------------------------------------------
__global__ __launch_bounds__(256) void k_bound(const unsigned char* __restrict__ g,
                                               const int* __restrict__ lastOut,
                                               int* __restrict__ boundary) {
    __shared__ unsigned char gS[NCHUNK * 32];
    const uint4* src = (const uint4*)g;
    uint4* dst = (uint4*)gS;
    for (int t = threadIdx.x; t < NCHUNK * 32 / 16; t += blockDim.x) dst[t] = src[t];
    __syncthreads();
    if (threadIdx.x == 0) {
        int b = *lastOut;
        boundary[NCHUNK] = b;
        for (int c = NCHUNK - 1; c >= 0; --c) {
            b = (int)gS[c * 32 + b];
            boundary[c] = b;
        }
    }
}

// ---------------------------------------------------------------------------
// K6: per-chunk backtrack with the known end label; bp tile staged into LDS
// cooperatively so the dependent walk is LDS-latency. Writes path[beg..end-1].
// ---------------------------------------------------------------------------
__global__ __launch_bounds__(32) void k_path(const unsigned char* __restrict__ bp,
                                             const int* __restrict__ boundary,
                                             int* __restrict__ path) {
    const int c = blockIdx.x;
    const long beg = (long)c * CHUNKL;
    long end = beg + CHUNKL; if (end > MSEQ - 1) end = MSEQ - 1;
    if (beg >= MSEQ - 1) return;  // empty chunk

    __shared__ unsigned char bpS[CHUNKL * 32];
    const int n16 = (int)((end - beg) * 32 / 16);
    const uint4* src = (const uint4*)(bp + (size_t)(beg + 1) * 32);
    uint4* dst = (uint4*)bpS;
    for (int t = threadIdx.x; t < n16; t += 32) dst[t] = src[t];
    __syncthreads();

    if (threadIdx.x == 0) {
        int curl = boundary[c + 1];
        for (long i = end; i >= beg + 1; --i) {
            curl = (int)bpS[(i - beg - 1) * 32 + curl];
            path[i - 1] = curl;
        }
    }
}

// ---------------------------------------------------------------------------
extern "C" void kernel_launch(void* const* d_in, const int* in_sizes, int n_in,
                              void* d_out, int out_size, void* d_ws, size_t ws_size,
                              hipStream_t stream) {
    const float* x = (const float*)d_in[0];   // [M, 128]
    const float* w = (const float*)d_in[1];   // [26, 128]
    const float* T = (const float*)d_in[2];   // [26, 26]
    int* path = (int*)d_out;                  // [M] int32 labels

    // workspace carve-up (all 16B aligned except lastOut)
    char* ws = (char*)d_ws;
    constexpr size_t SC_B  = (size_t)MSEQ * 32 * sizeof(float);            // 25.6 MB
    constexpr size_t RG_B  = (size_t)NCHUNK * KLAB * KLAB * sizeof(float); // 2.77 MB
    constexpr size_t AI_B  = (size_t)NCHUNK * 32 * sizeof(float);          // 128 KB
    constexpr size_t BP_B  = (size_t)MSEQ * 32;                            // 6.4 MB
    constexpr size_t G_B   = (size_t)NCHUNK * 32;                          // 32 KB
    constexpr size_t BD_B  = (size_t)(NCHUNK + 2) * sizeof(int);

    float*         scores   = (float*)(ws);
    float*         Rg       = (float*)(ws + SC_B);
    float*         alphaIn  = (float*)(ws + SC_B + RG_B);
    unsigned char* bp       = (unsigned char*)(ws + SC_B + RG_B + AI_B);
    unsigned char* g        = (unsigned char*)(ws + SC_B + RG_B + AI_B + BP_B);
    int*           boundary = (int*)(ws + SC_B + RG_B + AI_B + BP_B + G_B);
    int*           lastOut  = (int*)(ws + SC_B + RG_B + AI_B + BP_B + G_B + BD_B);
    (void)in_sizes; (void)n_in; (void)out_size; (void)ws_size;

    // K1: emission scores via WMMA (12500 row-tiles, 4 waves/block)
    k_scores<<<MSEQ / 64, 128, 0, stream>>>(x, w, scores);
    // K2: per-chunk max-plus matrices
    k_chunkmat<<<NCHUNK, 32, 0, stream>>>(scores, T, Rg);
    // K3: scan -> alpha at chunk starts + final label
    k_scan<<<1, 32, 0, stream>>>(scores, Rg, alphaIn, lastOut, path);
    // K4: backpointers + per-chunk end->start maps
    k_backptr<<<NCHUNK, 32, 0, stream>>>(scores, T, alphaIn, bp, g);
    // K5: boundary labels
    k_bound<<<1, 256, 0, stream>>>(g, lastOut, boundary);
    // K6: emit path
    k_path<<<NCHUNK, 32, 0, stream>>>(bp, boundary, path);
}
